// ShaunLSTM_35811437314406
// MI455X (gfx1250) — matrix operand order repaired and checked by reading, not compiled, not run on hardware
//
#include <hip/hip_runtime.h>

// CDNA5 / gfx1250 fused LSTM: 20 WGs x 16 rows.
// - weights resident in registers as bf16 WMMA B-fragments
// - v_wmma_f32_16x16x32_bf16 for both x@W and h@U
// - x slabs double-buffered in LDS via global_load_async_to_lds_b128 (ASYNCcnt),
//   removing global-load latency from the serial per-timestep chain.

typedef __attribute__((ext_vector_type(16))) __bf16 v16bf;
typedef __attribute__((ext_vector_type(8)))  float  v8f;

#define T_STEPS 2048
#define NB      320
#define F_IN    128
#define H_DIM   64
#define G4H     256

// padded LDS strides (floats / bf16 elems) to avoid bank conflicts
#define XS_LD 132
#define ZS_LD 260
#define HS_LD 68

__device__ __forceinline__ v8f wmma_bf16(v16bf a, v16bf b, v8f c) {
  return __builtin_amdgcn_wmma_f32_16x16x32_bf16(
      /*neg_a=*/false, a, /*neg_b=*/false, b,
      /*c_mod=*/(short)0, c, /*reuse_a=*/false, /*reuse_b=*/false);
}

__device__ __forceinline__ float fast_sigmoid(float x) {
  return __builtin_amdgcn_rcpf(1.0f + __expf(-x));
}
__device__ __forceinline__ float fast_tanh(float x) {
  return 2.0f * __builtin_amdgcn_rcpf(1.0f + __expf(-2.0f * x)) - 1.0f;
}

// Async DMA: 16 bytes per lane, global -> LDS, tracked by ASYNCcnt (GVS mode).
__device__ __forceinline__ void async_copy_b128(unsigned lds_off, unsigned voff,
                                                const float* base) {
  asm volatile("global_load_async_to_lds_b128 %0, %1, %2"
               :: "v"(lds_off), "v"(voff),
                  "s"((unsigned long long)(uintptr_t)base)
               : "memory");
}

// A-fragment (16x32 bf16, M x K) built from fp32 LDS tile, converting to bf16.
// ISA layout: lanes 0-15 = M, lane>>4 selects K-octet; v0..3 -> K in [kh*8, kh*8+8),
// v4..7 -> K in [16+kh*8, 16+kh*8+8), 2 bf16 per VGPR.
__device__ __forceinline__ v16bf load_a_f32(const float* s, int ld, int k0, int lane) {
  const int m  = lane & 15;
  const int kh = lane >> 4;
  const float* row = s + m * ld;
  v16bf a;
#pragma unroll
  for (int v = 0; v < 8; ++v) {
    const int k = k0 + ((v & 4) << 2) + kh * 8 + ((v & 3) << 1);
    a[2 * v]     = (__bf16)row[k];
    a[2 * v + 1] = (__bf16)row[k + 1];
  }
  return a;
}

__device__ __forceinline__ v16bf load_a_bf16(const __bf16* s, int ld, int k0, int lane) {
  const int m  = lane & 15;
  const int kh = lane >> 4;
  const __bf16* row = s + m * ld;
  v16bf a;
#pragma unroll
  for (int v = 0; v < 8; ++v) {
    const int k = k0 + ((v & 4) << 2) + kh * 8 + ((v & 3) << 1);
    a[2 * v]     = row[k];
    a[2 * v + 1] = row[k + 1];
  }
  return a;
}

// B-fragment (32x16 bf16, K x N) loaded once from global fp32 weights.
// ISA layout: lane&15 = N, lane>>4 selects K-octet; v0..3 -> K in [kh*8, kh*8+8), v4..7 -> K in [16+kh*8, 16+kh*8+8).
__device__ __forceinline__ v16bf load_b_global(const float* w, int k0, int n0, int lane) {
  const int n  = n0 + (lane & 15);
  const int kh = lane >> 4;
  v16bf b;
#pragma unroll
  for (int v = 0; v < 8; ++v) {
    const int k = k0 + ((v & 4) << 2) + kh * 8 + ((v & 3) << 1);
    b[2 * v]     = (__bf16)w[(size_t)k * G4H + n];
    b[2 * v + 1] = (__bf16)w[(size_t)(k + 1) * G4H + n];
  }
  return b;
}

__global__ __launch_bounds__(256)
void ShaunLSTM_35811437314406_kernel(const float* __restrict__ x,
                                     const float* __restrict__ W,
                                     const float* __restrict__ U,
                                     const float* __restrict__ b,
                                     float* __restrict__ out) {
  __shared__ float  xs[2][16 * XS_LD]; // double-buffered x slab, fp32
  __shared__ float  zs[16 * ZS_LD];    // gate pre-activations, fp32
  __shared__ __bf16 hs[16 * HS_LD];    // hidden state, bf16 (A-operand source)

  const int tid  = threadIdx.x;
  const int lane = tid & 31;
  const int wave = tid >> 5;           // 0..7, wave w owns N-tiles 2w, 2w+1
  const int r0   = blockIdx.x * 16;    // batch-row base

  // per-thread async-copy addressing: thread covers elements [tid*4, +4) and
  // [(tid+256)*4, +4) of each 2048-float slab; padded row layout in LDS.
  const int e0 = tid << 2, e1 = (tid + 256) << 2;
  const unsigned ldsA[2] = {
      (unsigned)(uintptr_t)&xs[0][(e0 >> 7) * XS_LD + (e0 & 127)],
      (unsigned)(uintptr_t)&xs[1][(e0 >> 7) * XS_LD + (e0 & 127)]};
  const unsigned ldsB[2] = {
      (unsigned)(uintptr_t)&xs[0][(e1 >> 7) * XS_LD + (e1 & 127)],
      (unsigned)(uintptr_t)&xs[1][(e1 >> 7) * XS_LD + (e1 & 127)]};
  const unsigned voff0 = (unsigned)(tid * 16);
  const unsigned voff1 = (unsigned)(tid * 16 + 4096);

  // ---- prologue: kick off slab 0 DMA, then set up resident state under it ----
  {
    const float* base0 = x + ((size_t)0 * NB + r0) * F_IN;
    async_copy_b128(ldsA[0], voff0, base0);
    async_copy_b128(ldsB[0], voff1, base0);
  }

  // resident weights: bf16 B-fragments in registers (loaded once)
  v16bf Wfrag[4][2];                   // K-blocks 0..3 of F_IN=128
  v16bf Ufrag[2][2];                   // K-blocks 0..1 of H_DIM=64
#pragma unroll
  for (int i = 0; i < 2; ++i) {
    const int n0 = (2 * wave + i) * 16;
#pragma unroll
    for (int kt = 0; kt < 4; ++kt) Wfrag[kt][i] = load_b_global(W, kt * 32, n0, lane);
#pragma unroll
    for (int kt = 0; kt < 2; ++kt) Ufrag[kt][i] = load_b_global(U, kt * 32, n0, lane);
  }

  // elementwise-phase constants: thread owns column j, rows mb+4k
  const int   j  = tid & 63;
  const int   mb = tid >> 6;           // 0..3
  const float bi = b[j], bff = b[64 + j], bg = b[128 + j], bo = b[192 + j];
  float cst[4] = {0.f, 0.f, 0.f, 0.f};

  for (int idx = tid; idx < 16 * HS_LD; idx += 256) hs[idx] = (__bf16)0.f;

  asm volatile("s_wait_asynccnt 0x0" ::: "memory");
  __syncthreads();

#pragma unroll 1
  for (int t = 0; t < T_STEPS; ++t) {
    // ---- phase 1: kick DMA for slab t+1, drain slab t, one barrier ----
    if (t + 1 < T_STEPS) {
      const int nb = (t + 1) & 1;
      const float* nbase = x + ((size_t)(t + 1) * NB + r0) * F_IN;
      async_copy_b128(ldsA[nb], voff0, nbase);
      async_copy_b128(ldsB[nb], voff1, nbase);
      // in-order completion: <=2 outstanding => slab t has fully landed
      asm volatile("s_wait_asynccnt 0x2" ::: "memory");
    } else {
      asm volatile("s_wait_asynccnt 0x0" ::: "memory");
    }
    __syncthreads();  // xs[t&1] visible from all waves; hs from step t-1 visible

    const float* xbuf = xs[t & 1];

    // ---- phase 2: z = h@U (critical path first) + x@W, two accumulators ----
    v8f acc_h[2]; acc_h[0] = {}; acc_h[1] = {};
    v8f acc_x[2]; acc_x[0] = {}; acc_x[1] = {};
#pragma unroll
    for (int kt = 0; kt < 2; ++kt) {
      const v16bf a = load_a_bf16(hs, HS_LD, kt * 32, lane);
      acc_h[0] = wmma_bf16(a, Ufrag[kt][0], acc_h[0]);
      acc_h[1] = wmma_bf16(a, Ufrag[kt][1], acc_h[1]);
    }
#pragma unroll
    for (int kt = 0; kt < 4; ++kt) {
      const v16bf a = load_a_f32(xbuf, XS_LD, kt * 32, lane);
      acc_x[0] = wmma_bf16(a, Wfrag[kt][0], acc_x[0]);
      acc_x[1] = wmma_bf16(a, Wfrag[kt][1], acc_x[1]);
    }
    {
      const int col0 = (2 * wave) * 16 + (lane & 15);
      const int rb   = (lane >> 4) * 8;       // C/D layout: half-wave = M+8
#pragma unroll
      for (int r = 0; r < 8; ++r) {
        zs[(rb + r) * ZS_LD + col0]      = acc_h[0][r] + acc_x[0][r];
        zs[(rb + r) * ZS_LD + col0 + 16] = acc_h[1][r] + acc_x[1][r];
      }
    }
    __syncthreads();

    // ---- phase 3: gates, cell/hidden update, write-out ----
#pragma unroll
    for (int k = 0; k < 4; ++k) {
      const int   m  = mb + 4 * k;
      const float zi = zs[m * ZS_LD + j]        + bi;
      const float zf = zs[m * ZS_LD + 64 + j]   + bff;
      const float zg = zs[m * ZS_LD + 128 + j]  + bg;
      const float zo = zs[m * ZS_LD + 192 + j]  + bo;
      const float ig = fast_sigmoid(zi);
      const float fg = fast_sigmoid(zf);
      const float gg = fast_tanh(zg);
      const float og = fast_sigmoid(zo);
      const float c  = fg * cst[k] + ig * gg;
      cst[k] = c;
      const float h = og * fast_tanh(c);
      out[((size_t)t * NB + (r0 + m)) * H_DIM + j] = h;
      hs[m * HS_LD + j] = (__bf16)h;
    }
    // no barrier here: next iteration's single barrier orders hs/zs/xs reuse
  }
}

extern "C" void kernel_launch(void* const* d_in, const int* in_sizes, int n_in,
                              void* d_out, int out_size, void* d_ws, size_t ws_size,
                              hipStream_t stream) {
  (void)in_sizes; (void)n_in; (void)d_ws; (void)ws_size; (void)out_size;
  const float* x = (const float*)d_in[0];
  const float* W = (const float*)d_in[1];
  const float* U = (const float*)d_in[2];
  const float* b = (const float*)d_in[3];
  float* out = (float*)d_out;
  ShaunLSTM_35811437314406_kernel<<<NB / 16, 256, 0, stream>>>(x, W, U, b, out);
}
